// BiMambaBlock_88519275970819
// MI455X (gfx1250) — matrix operand order repaired and checked by reading, not compiled, not run on hardware
//
#include <hip/hip_runtime.h>

// Problem constants (match reference)
#define BB    2
#define LL    1024
#define DM    768
#define DI    1536
#define DSN   16
#define DTR   48
#define DTRP  64          // dt_w K padded to 64 (zero-filled) so K%64==0
#define NXP   80          // DT_RANK + 2*D_STATE
#define BLROWS (BB*LL)    // 2048

typedef __attribute__((ext_vector_type(4)))  unsigned int v4u;
typedef __attribute__((ext_vector_type(8)))  unsigned int v8u;
typedef __attribute__((ext_vector_type(16))) __bf16       v16bf;
typedef __attribute__((ext_vector_type(8)))  float        v8f;

__device__ __forceinline__ unsigned short f2bf(float f) {
  unsigned int u = __float_as_uint(f);
  u += 0x7FFFu + ((u >> 16) & 1u);       // round-to-nearest-even
  return (unsigned short)(u >> 16);
}

__device__ __forceinline__ v16bf mkfrag(v4u lo, v4u hi) {
  v8u u;
  u[0] = lo[0]; u[1] = lo[1]; u[2] = lo[2]; u[3] = lo[3];
  u[4] = hi[0]; u[5] = hi[1]; u[6] = hi[2]; u[7] = hi[3];
  return __builtin_bit_cast(v16bf, u);
}

__device__ __forceinline__ float epilogue_act(float f, int mode) {
  if (mode == 1) {                       // softplus
    f = (f > 20.f) ? f : log1pf(__expf(f));
  } else if (mode == 2) {                // exact gelu
    f = 0.5f * f * (1.f + erff(f * 0.70710678118654752f));
  }
  return f;
}

// ---------------------------------------------------------------------------
// f32 -> bf16 convert
// ---------------------------------------------------------------------------
__global__ void cvt_bf16_kernel(const float* __restrict__ in,
                                unsigned short* __restrict__ out, int n) {
  int i = blockIdx.x * blockDim.x + threadIdx.x;
  if (i < n) out[i] = f2bf(in[i]);
}

// f32 (rows x inK) -> bf16 (rows x outK) with zero padding of cols inK..outK
__global__ void cvt_pad_bf16_kernel(const float* __restrict__ in,
                                    unsigned short* __restrict__ out,
                                    int rows, int inK, int outK) {
  int i = blockIdx.x * blockDim.x + threadIdx.x;
  if (i >= rows * outK) return;
  int r = i / outK, k = i % outK;
  out[i] = (k < inK) ? f2bf(in[r * inK + k]) : (unsigned short)0;
}

// ---------------------------------------------------------------------------
// LayerNorm over D columns; writes bf16 output and (optionally) a copy with
// the time axis flipped within each batch (for the backward mamba direction).
// ---------------------------------------------------------------------------
__global__ void ln_kernel(const float* __restrict__ x,
                          const float* __restrict__ g,
                          const float* __restrict__ b,
                          unsigned short* __restrict__ out_bf,
                          unsigned short* __restrict__ out_rev_bf,
                          int Lr, int D) {
  int row = blockIdx.x;                // [0, B*L)
  int bat = row / Lr, t = row % Lr;
  const float* xr = x + (size_t)row * D;

  __shared__ float s1[256];
  __shared__ float s2[256];
  float lsum = 0.f, lsq = 0.f;
  for (int c = threadIdx.x; c < D; c += 256) {
    float v = xr[c];
    lsum += v; lsq += v * v;
  }
  s1[threadIdx.x] = lsum; s2[threadIdx.x] = lsq;
  __syncthreads();
  for (int s = 128; s > 0; s >>= 1) {
    if ((int)threadIdx.x < s) {
      s1[threadIdx.x] += s1[threadIdx.x + s];
      s2[threadIdx.x] += s2[threadIdx.x + s];
    }
    __syncthreads();
  }
  float mean = s1[0] / (float)D;
  float var  = s2[0] / (float)D - mean * mean;
  float inv  = rsqrtf(var + 1e-5f);

  size_t rrev = (size_t)bat * Lr + (Lr - 1 - t);
  for (int c = threadIdx.x; c < D; c += 256) {
    float v = (xr[c] - mean) * inv * g[c] + b[c];
    unsigned short h = f2bf(v);
    out_bf[(size_t)row * D + c] = h;
    if (out_rev_bf) out_rev_bf[rrev * D + c] = h;
  }
}

// ---------------------------------------------------------------------------
// bf16 WMMA GEMM:  C[M,N] = act( A[M,K] @ W[N,K]^T + bias[N] )
// Requirements: K % 64 == 0, M % 16 == 0, N % 16 == 0, rows 16B-aligned.
// Each wave computes a 16x32 tile (two 16x16 n-tiles sharing the A fragment)
// with K unrolled x2 -> 4 independent WMMA accumulator chains.
// Fragment layouts (C/D per CDNA5 ISA 7.12.2; A/B use this code's contiguous k order):
//   A (16x32 bf16): lane=row%16; K = lanehalf*16 + {0..15}        (2x b128)
//   B (32x16 bf16): lane=col%16; K = lanehalf*16 + {0..15}        (2x b128)
//   C/D (16x16 f32): n = lane%16, m = vgpr + 8*lanehalf
// mode: 0 = none, 1 = softplus, 2 = exact gelu.  Cb (optional) = bf16 mirror.
// ---------------------------------------------------------------------------
__global__ void __launch_bounds__(256)
wmma_gemm_kernel(const unsigned short* __restrict__ A, int lda,
                 const unsigned short* __restrict__ W, int ldw,
                 float* __restrict__ C, int ldc,
                 unsigned short* __restrict__ Cb,
                 const float* __restrict__ bias,
                 int M, int N, int K, int mode) {
  const int wave = threadIdx.x >> 5;
  const int lane = threadIdx.x & 31;
  const int lh = lane >> 4;            // lane half (0/1)
  const int ln = lane & 15;

  const int mtile = blockIdx.y * 2 + (wave >> 2);
  const int nt0   = (blockIdx.x * 4 + (wave & 3)) * 2;
  if (mtile * 16 >= M || nt0 * 16 >= N) return;   // uniform per wave

  const bool hasN1 = (nt0 + 1) * 16 < N;          // uniform per wave
  const int n0 = nt0 * 16 + ln;
  const int n1 = hasN1 ? n0 + 16 : n0;

  const unsigned short* arow = A + (size_t)(mtile * 16 + ln) * lda + lh * 16;
  const unsigned short* w0   = W + (size_t)n0 * ldw + lh * 16;
  const unsigned short* w1   = W + (size_t)n1 * ldw + lh * 16;

  v8f acc0a = 0.f, acc0b = 0.f, acc1a = 0.f, acc1b = 0.f;

  for (int kt = 0; kt < K; kt += 64) {
    __builtin_prefetch(arow + kt + 128, 0, 3);
    __builtin_prefetch(w0 + kt + 128, 0, 3);
    __builtin_prefetch(w1 + kt + 128, 0, 3);

    v16bf aA  = mkfrag(*(const v4u*)(arow + kt),      *(const v4u*)(arow + kt + 8));
    v16bf aB  = mkfrag(*(const v4u*)(arow + kt + 32), *(const v4u*)(arow + kt + 40));
    v16bf b0A = mkfrag(*(const v4u*)(w0 + kt),        *(const v4u*)(w0 + kt + 8));
    v16bf b0B = mkfrag(*(const v4u*)(w0 + kt + 32),   *(const v4u*)(w0 + kt + 40));
    v16bf b1A = mkfrag(*(const v4u*)(w1 + kt),        *(const v4u*)(w1 + kt + 8));
    v16bf b1B = mkfrag(*(const v4u*)(w1 + kt + 32),   *(const v4u*)(w1 + kt + 40));

    acc0a = __builtin_amdgcn_wmma_f32_16x16x32_bf16(false, aA, false, b0A,
                                                    (short)0, acc0a, false, false);
    acc1a = __builtin_amdgcn_wmma_f32_16x16x32_bf16(false, aA, false, b1A,
                                                    (short)0, acc1a, false, false);
    acc0b = __builtin_amdgcn_wmma_f32_16x16x32_bf16(false, aB, false, b0B,
                                                    (short)0, acc0b, false, false);
    acc1b = __builtin_amdgcn_wmma_f32_16x16x32_bf16(false, aB, false, b1B,
                                                    (short)0, acc1b, false, false);
  }

  const float bia0 = bias ? bias[n0] : 0.f;
  const float bia1 = bias ? bias[n1] : 0.f;
#pragma unroll
  for (int r = 0; r < 8; ++r) {
    int m = mtile * 16 + lh * 8 + r;
    size_t rowo = (size_t)m * ldc;
    float f0 = epilogue_act(acc0a[r] + acc0b[r] + bia0, mode);
    C[rowo + n0] = f0;
    if (Cb) Cb[rowo + n0] = f2bf(f0);
    if (hasN1) {
      float f1 = epilogue_act(acc1a[r] + acc1b[r] + bia1, mode);
      C[rowo + n1] = f1;
      if (Cb) Cb[rowo + n1] = f2bf(f1);
    }
  }
}

// ---------------------------------------------------------------------------
// Depthwise causal conv (width 4) + SiLU on the xc half of xz.
// xz rows are (B*L, 2*DI); xc = cols [0,DI).
// ---------------------------------------------------------------------------
__global__ void conv_silu_kernel(const float* __restrict__ xz,
                                 const float* __restrict__ cw,   // (DI,4)
                                 const float* __restrict__ cb,   // (DI)
                                 float* __restrict__ xc,
                                 unsigned short* __restrict__ xcbf,
                                 int Lr) {
  size_t i = (size_t)blockIdx.x * blockDim.x + threadIdx.x;
  size_t total = (size_t)BB * Lr * DI;
  if (i >= total) return;
  int d = (int)(i % DI);
  size_t r = i / DI;
  int bat = (int)(r / Lr), t = (int)(r % Lr);

  float s = cb[d];
#pragma unroll
  for (int j = 0; j < 4; ++j) {
    int tt = t - 3 + j;
    if (tt >= 0)
      s += xz[((size_t)bat * Lr + tt) * (2 * DI) + d] * cw[d * 4 + j];
  }
  s = s / (1.f + __expf(-s));          // SiLU
  xc[i] = s;
  xcbf[i] = f2bf(s);
}

// ---------------------------------------------------------------------------
// Selective-scan: sequential over t, one thread per (batch, channel d).
// Fuses the D-skip and the SiLU(z) gate; emits bf16 for the output GEMM.
// ---------------------------------------------------------------------------
__global__ void scan_kernel(const float* __restrict__ delta,  // (BL, DI)
                            const float* __restrict__ dbc,    // (BL, 80): B@48, C@64
                            const float* __restrict__ xc,     // (BL, DI)
                            const float* __restrict__ xz,     // (BL, 2*DI): z @ +DI
                            const float* __restrict__ A_log,  // (DI, 16)
                            const float* __restrict__ Dskip,  // (DI)
                            unsigned short* __restrict__ ybf, // (BL, DI)
                            int Lr) {
  int tid = blockIdx.x * blockDim.x + threadIdx.x;
  if (tid >= BB * DI) return;
  int bat = tid / DI;
  int d   = tid % DI;

  float As[DSN];
#pragma unroll
  for (int s = 0; s < DSN; ++s) As[s] = -__expf(A_log[d * DSN + s]);
  const float Dv = Dskip[d];

  float h[DSN];
#pragma unroll
  for (int s = 0; s < DSN; ++s) h[s] = 0.f;

  for (int t = 0; t < Lr; ++t) {
    size_t r = (size_t)bat * Lr + t;
    float dl = delta[r * DI + d];
    float xv = xc[r * DI + d];
    float dx = dl * xv;
    float acc = 0.f;
#pragma unroll
    for (int s = 0; s < DSN; ++s) {
      float Bs = dbc[r * NXP + DTR + s];
      float Cs = dbc[r * NXP + DTR + DSN + s];
      h[s] = __expf(dl * As[s]) * h[s] + dx * Bs;
      acc += h[s] * Cs;
    }
    float y = acc + xv * Dv;
    float zv = xz[r * (2 * DI) + DI + d];
    y *= zv / (1.f + __expf(-zv));     // gate by SiLU(z)
    ybf[r * DI + d] = f2bf(y);
  }
}

// ---------------------------------------------------------------------------
// xmid = x + 0.5*(y_fwd + flip_t(y_bwd))
// ---------------------------------------------------------------------------
__global__ void combine_kernel(const float* __restrict__ x,
                               const float* __restrict__ yf,
                               const float* __restrict__ yb,
                               float* __restrict__ xmid, int Lr) {
  size_t i = (size_t)blockIdx.x * blockDim.x + threadIdx.x;
  size_t total = (size_t)BB * Lr * DM;
  if (i >= total) return;
  int d = (int)(i % DM);
  size_t r = i / DM;
  int bat = (int)(r / Lr), t = (int)(r % Lr);
  size_t rrev = (size_t)bat * Lr + (Lr - 1 - t);
  xmid[i] = x[i] + 0.5f * (yf[i] + yb[rrev * DM + d]);
}

__global__ void add_kernel(const float* __restrict__ a,
                           const float* __restrict__ b,
                           float* __restrict__ out, size_t n) {
  size_t i = (size_t)blockIdx.x * blockDim.x + threadIdx.x;
  if (i < n) out[i] = a[i] + b[i];
}

// ---------------------------------------------------------------------------
// Host side
// ---------------------------------------------------------------------------
static void launch_gemm(hipStream_t s,
                        const unsigned short* A, int lda,
                        const unsigned short* W, int ldw,
                        float* C, int ldc, unsigned short* Cb,
                        const float* bias, int M, int N, int K, int mode) {
  dim3 grid((N + 127) / 128, (M + 31) / 32);
  wmma_gemm_kernel<<<grid, 256, 0, s>>>(A, lda, W, ldw, C, ldc, Cb, bias,
                                        M, N, K, mode);
}

static void launch_cvt(hipStream_t s, const float* in, unsigned short* out, int n) {
  cvt_bf16_kernel<<<(n + 255) / 256, 256, 0, s>>>(in, out, n);
}

extern "C" void kernel_launch(void* const* d_in, const int* in_sizes, int n_in,
                              void* d_out, int out_size, void* d_ws, size_t ws_size,
                              hipStream_t stream) {
  (void)in_sizes; (void)n_in; (void)out_size; (void)ws_size;

  const float* x      = (const float*)d_in[0];
  const float* ln_g   = (const float*)d_in[1];
  const float* ln_b   = (const float*)d_in[2];
  // per-direction params: base index 3 (fwd) / 12 (bwd)
  const float* ffn_g  = (const float*)d_in[21];
  const float* ffn_b  = (const float*)d_in[22];
  const float* w1     = (const float*)d_in[23];
  const float* b1     = (const float*)d_in[24];
  const float* w2     = (const float*)d_in[25];
  const float* b2     = (const float*)d_in[26];
  float* out          = (float*)d_out;

  const int N_INW   = 2 * DI * DM;   // 2359296
  const int N_XPW   = NXP * DI;      // 122880
  const int N_DTWP  = DI * DTRP;     // 98304 (padded)
  const int N_OUTW  = DM * DI;       // 1179648
  const int N_W1    = 4 * DM * DM;   // 2359296
  const int N_W2    = DM * 4 * DM;   // 2359296
  const size_t BLD  = (size_t)BLROWS * DM;      // 1572864
  const size_t BLDI = (size_t)BLROWS * DI;      // 3145728
  const size_t BLXZ = (size_t)BLROWS * 2 * DI;  // 6291456
  const size_t BLP  = (size_t)BLROWS * NXP;     // 163840

  // bump allocator over d_ws
  size_t off = 0;
  auto alloc = [&](size_t bytes) -> void* {
    off = (off + 255) & ~(size_t)255;
    void* p = (char*)d_ws + off;
    off += bytes;
    return p;
  };

  unsigned short* inw_bf[2]  = { (unsigned short*)alloc((size_t)N_INW * 2),
                                 (unsigned short*)alloc((size_t)N_INW * 2) };
  unsigned short* xpw_bf[2]  = { (unsigned short*)alloc((size_t)N_XPW * 2),
                                 (unsigned short*)alloc((size_t)N_XPW * 2) };
  unsigned short* dtw_bf[2]  = { (unsigned short*)alloc((size_t)N_DTWP * 2),
                                 (unsigned short*)alloc((size_t)N_DTWP * 2) };
  unsigned short* outw_bf[2] = { (unsigned short*)alloc((size_t)N_OUTW * 2),
                                 (unsigned short*)alloc((size_t)N_OUTW * 2) };
  unsigned short* w1_bf  = (unsigned short*)alloc((size_t)N_W1 * 2);
  unsigned short* w2_bf  = (unsigned short*)alloc((size_t)N_W2 * 2);

  unsigned short* xn_bf    = (unsigned short*)alloc(BLD * 2);
  unsigned short* xnrev_bf = (unsigned short*)alloc(BLD * 2);
  float*          xz       = (float*)alloc(BLXZ * 4);     // reused as h1_f32
  float*          xc       = (float*)alloc(BLDI * 4);
  unsigned short* xc_bf    = (unsigned short*)alloc(BLDI * 2);
  float*          dbc      = (float*)alloc(BLP * 4);
  unsigned short* dbc_bf   = (unsigned short*)alloc(BLP * 2);
  float*          delta    = (float*)alloc(BLDI * 4);
  unsigned short* y_bf     = (unsigned short*)alloc(BLDI * 2);
  float*          yproj[2] = { (float*)alloc(BLD * 4), (float*)alloc(BLD * 4) };
  float*          xmid     = (float*)alloc(BLD * 4);
  unsigned short* h_bf     = (unsigned short*)alloc(BLD * 2);
  unsigned short* h1_bf    = (unsigned short*)alloc(BLXZ * 2);
  float*          h2       = (float*)alloc(BLD * 4);

  // ---- weight conversion (deterministic each call) ----
  for (int dir = 0; dir < 2; ++dir) {
    int base = 3 + dir * 9;
    launch_cvt(stream, (const float*)d_in[base + 0], inw_bf[dir],  N_INW);
    launch_cvt(stream, (const float*)d_in[base + 3], xpw_bf[dir],  N_XPW);
    // dt_w: (DI x 48) -> (DI x 64) zero-padded so the GEMM needs no K guard
    cvt_pad_bf16_kernel<<<(N_DTWP + 255) / 256, 256, 0, stream>>>(
        (const float*)d_in[base + 4], dtw_bf[dir], DI, DTR, DTRP);
    launch_cvt(stream, (const float*)d_in[base + 8], outw_bf[dir], N_OUTW);
  }
  launch_cvt(stream, w1, w1_bf, N_W1);
  launch_cvt(stream, w2, w2_bf, N_W2);

  // ---- input LayerNorm (+ flipped copy for backward direction) ----
  ln_kernel<<<BLROWS, 256, 0, stream>>>(x, ln_g, ln_b, xn_bf, xnrev_bf, LL, DM);

  // ---- two mamba directions ----
  for (int dir = 0; dir < 2; ++dir) {
    int base = 3 + dir * 9;
    const float* conv_w = (const float*)d_in[base + 1];
    const float* conv_b = (const float*)d_in[base + 2];
    const float* dt_b   = (const float*)d_in[base + 5];
    const float* A_log  = (const float*)d_in[base + 6];
    const float* Dskip  = (const float*)d_in[base + 7];
    const unsigned short* src = dir ? xnrev_bf : xn_bf;

    // xz = xn @ in_w.T            (2048 x 3072 x 768)
    launch_gemm(stream, src, DM, inw_bf[dir], DM, xz, 2 * DI, nullptr, nullptr,
                BLROWS, 2 * DI, DM, 0);
    // depthwise conv + SiLU
    conv_silu_kernel<<<(int)((BLDI + 255) / 256), 256, 0, stream>>>(
        xz, conv_w, conv_b, xc, xc_bf, LL);
    // dbc = xc @ xp_w.T           (2048 x 80 x 1536)
    launch_gemm(stream, xc_bf, DI, xpw_bf[dir], DI, dbc, NXP, dbc_bf, nullptr,
                BLROWS, NXP, DI, 0);
    // delta = softplus(dt @ dt_w.T + dt_b)
    // A = dbc_bf (lda=80), K=64: cols 48..63 hit zero-padded weight cols -> 0
    launch_gemm(stream, dbc_bf, NXP, dtw_bf[dir], DTRP, delta, DI, nullptr, dt_b,
                BLROWS, DI, DTRP, 1);
    // selective scan (fused D-skip + SiLU(z) gate) -> bf16
    scan_kernel<<<(BB * DI + 255) / 256, 256, 0, stream>>>(
        delta, dbc, xc, xz, A_log, Dskip, y_bf, LL);
    // y @ out_w.T                 (2048 x 768 x 1536)
    launch_gemm(stream, y_bf, DI, outw_bf[dir], DI, yproj[dir], DM, nullptr,
                nullptr, BLROWS, DM, DI, 0);
  }

  // ---- combine directions + residual ----
  combine_kernel<<<(int)((BLD + 255) / 256), 256, 0, stream>>>(
      x, yproj[0], yproj[1], xmid, LL);

  // ---- FFN ----
  ln_kernel<<<BLROWS, 256, 0, stream>>>(xmid, ffn_g, ffn_b, h_bf, nullptr, LL, DM);
  // h1 = gelu(h @ w1.T + b1)      (2048 x 3072 x 768) ; reuse xz as f32 out
  launch_gemm(stream, h_bf, DM, w1_bf, DM, xz, 4 * DM, h1_bf, b1,
              BLROWS, 4 * DM, DM, 2);
  // h2 = h1 @ w2.T + b2           (2048 x 768 x 3072)
  launch_gemm(stream, h1_bf, 4 * DM, w2_bf, 4 * DM, h2, DM, nullptr, b2,
              BLROWS, DM, 4 * DM, 0);
  // out = xmid + h2
  add_kernel<<<(int)((BLD + 255) / 256), 256, 0, stream>>>(xmid, h2, out, BLD);
}
